// Codebook_39814346834091
// MI455X (gfx1250) — matrix operand order repaired and checked by reading, not compiled, not run on hardware
//
#include <hip/hip_runtime.h>

typedef __attribute__((ext_vector_type(16))) _Float16 v16h;
typedef __attribute__((ext_vector_type(8)))  _Float16 v8h;
typedef __attribute__((ext_vector_type(8)))  float    v8f;

#define VQ_N     32768      // rows (32*1024)
#define VQ_D     256        // latent dim
#define VQ_V     8192       // vocab
#define VQ_S     4096.0f    // emb scale into f16 normal range
#define VQ_BETA  0.25f

#define WG_THREADS   256    // 8 waves
#define ROWS_PER_WAVE 32    // 2 M-tiles of 16
#define ROWS_PER_WG  (8 * ROWS_PER_WAVE)          // 256
#define MAIN_GRID    (VQ_N / ROWS_PER_WG)         // 128
#define N_PARTIALS   (MAIN_GRID * 8)              // 1024 wave partials
#define N_ITERS      (VQ_V / 16)                  // 512 code tiles

// LDS B tile: 16 codes x 256 K halfs, row stride padded to 264 halfs (528B)
// -> bank stride 132 dwords = 4 banks per row: conflict-free 16-lane reads.
#define TILE_STRIDE 264

static __device__ inline v8h cvt8(float4 a, float4 b, float s) {
    v8h h;
    h[0] = (_Float16)(a.x * s); h[1] = (_Float16)(a.y * s);
    h[2] = (_Float16)(a.z * s); h[3] = (_Float16)(a.w * s);
    h[4] = (_Float16)(b.x * s); h[5] = (_Float16)(b.y * s);
    h[6] = (_Float16)(b.z * s); h[7] = (_Float16)(b.w * s);
    return h;
}

static __device__ inline v16h ldB(const _Float16* p) {
    const v8h* bp = (const v8h*)p;
    v8h b0 = bp[0], b1 = bp[1];
    return __builtin_shufflevector(b0, b1, 0,1,2,3,4,5,6,7,8,9,10,11,12,13,14,15);
}

// ---------- kernel 1: e2s[v] = S * ||emb_v||^2 (exact f32) ----------
__global__ __launch_bounds__(256) void vq_e2_kernel(const float* __restrict__ emb,
                                                    float* __restrict__ e2s) {
    int wave = blockIdx.x * (blockDim.x >> 5) + (threadIdx.x >> 5);
    int lane = threadIdx.x & 31;
    const float4* p = (const float4*)(emb + (size_t)wave * VQ_D + lane * 8);
    float4 a = p[0], b = p[1];
    float s = a.x*a.x + a.y*a.y + a.z*a.z + a.w*a.w
            + b.x*b.x + b.y*b.y + b.z*b.z + b.w*b.w;
    #pragma unroll
    for (int off = 16; off >= 1; off >>= 1) s += __shfl_xor(s, off, 32);
    if (lane == 0) e2s[wave] = VQ_S * s;
}

// ---------- kernel 2: fused distance-matmul + argmin + gather + loss ----------
__global__ __launch_bounds__(WG_THREADS) void vq_main_kernel(
        const float* __restrict__ z, const float* __restrict__ emb,
        const float* __restrict__ e2s, float* __restrict__ zq,
        float* __restrict__ idx_out, float* __restrict__ partials) {
    __shared__ alignas(16) _Float16 tileB[16 * TILE_STRIDE];
    __shared__ float e2tile[16];

    const int tid  = threadIdx.x;
    const int wave = tid >> 5;
    const int lane = tid & 31;
    const int mr   = lane & 15;        // col (N) within tile / row within M-tile
    const int hi   = lane >> 4;        // lane half select
    const size_t waveRow = (size_t)blockIdx.x * ROWS_PER_WG + (size_t)wave * ROWS_PER_WAVE;

    // ---- load A fragments (z rows, f16) resident for whole kernel ----
    // 16-bit A 16x32 layout: elems 0..15 = K k0+16*hi.. (contiguous, matches B)
    v16h A[2][8];
    #pragma unroll
    for (int t = 0; t < 2; ++t) {
        const float* zrow = z + (waveRow + t * 16 + mr) * VQ_D;
        #pragma unroll
        for (int kt = 0; kt < 8; ++kt) {
            const float4* p0 = (const float4*)(zrow + kt * 32 + hi * 16);
            const float4* p1 = (const float4*)(zrow + kt * 32 + hi * 16 + 8);
            v8h lo = cvt8(p0[0], p0[1], 1.0f);
            v8h hih = cvt8(p1[0], p1[1], 1.0f);
            A[t][kt] = __builtin_shufflevector(lo, hih, 0,1,2,3,4,5,6,7,8,9,10,11,12,13,14,15);
        }
    }

    float minv[2][8];
    int   mini[2][8];
    #pragma unroll
    for (int t = 0; t < 2; ++t)
        #pragma unroll
        for (int r = 0; r < 8; ++r) { minv[t][r] = 3.402823466e+38f; mini[t][r] = 0; }

    const int stage_r   = tid >> 4;    // 0..15 code row
    const int stage_seg = tid & 15;    // 16-half segment
    v8h* const stage_dst = (v8h*)&tileB[stage_r * TILE_STRIDE + stage_seg * 16];
    // each thread stages exactly one 64B cacheline per iteration
    const char* const stage_src0 = (const char*)(emb + (size_t)stage_r * VQ_D + stage_seg * 16);
    const _Float16* const bbase = &tileB[mr * TILE_STRIDE + hi * 16];

    for (int it = 0; it < N_ITERS; ++it) {
        const int n0 = it << 4;
        __syncthreads();   // all waves done reading previous LDS tile
        // ---- stage tile `it` (cacheline was prefetched last iteration) ----
        {
            const float4* src = (const float4*)(stage_src0 + (size_t)n0 * (VQ_D * 4));
            v8h h0 = cvt8(src[0], src[1], VQ_S);
            v8h h1 = cvt8(src[2], src[3], VQ_S);
            stage_dst[0] = h0; stage_dst[1] = h1;
            if (tid < 16) e2tile[tid] = e2s[n0 + tid];
        }
        // ---- prefetch tile it+1 into cache: no VGPR cost, no LOADcnt, so the
        //      conservative barrier waits cannot serialize it ----
        if (it + 1 < N_ITERS)
            __builtin_prefetch(stage_src0 + (size_t)(n0 + 16) * (VQ_D * 4), 0, 3);
        __syncthreads();   // staged tile visible

        // ---- 16x16 tile x K=256: 8 wmma per M-tile, 2-deep B-fragment pipeline ----
        v8f acc0 = {}, acc1 = {};
        v16h bf0 = ldB(bbase);
        #pragma unroll
        for (int kt = 0; kt < 8; ++kt) {
            v16h bf1;
            if (kt + 1 < 8) bf1 = ldB(bbase + (kt + 1) * 32);   // issue ahead of kt's wait
            acc0 = __builtin_amdgcn_wmma_f32_16x16x32_f16(false, A[0][kt], false, bf0,
                                                          (short)0, acc0, false, false);
            acc1 = __builtin_amdgcn_wmma_f32_16x16x32_f16(false, A[1][kt], false, bf0,
                                                          (short)0, acc1, false, false);
            bf0 = bf1;
        }

        // score = S*(||e||^2 - 2 z.e): monotonic in true distance per row
        const float e2v = e2tile[mr];
        const int ncode = n0 + mr;
        #pragma unroll
        for (int r = 0; r < 8; ++r) {
            float s0 = e2v - 2.0f * acc0[r];
            float s1 = e2v - 2.0f * acc1[r];
            if (s0 < minv[0][r]) { minv[0][r] = s0; mini[0][r] = ncode; }
            if (s1 < minv[1][r]) { minv[1][r] = s1; mini[1][r] = ncode; }
        }
    }

    // ---- cross-lane argmin within 16-lane halves, then gather z_q + loss ----
    float lsum = 0.0f;
    #pragma unroll
    for (int t = 0; t < 2; ++t) {
        #pragma unroll
        for (int r = 0; r < 8; ++r) {
            float v = minv[t][r];
            int   i = mini[t][r];
            #pragma unroll
            for (int off = 1; off < 16; off <<= 1) {
                float ov = __shfl_xor(v, off, 32);
                int   oi = __shfl_xor(i, off, 32);
                if (ov < v || (ov == v && oi < i)) { v = ov; i = oi; }
            }
            // C/D layout: row = M-tile base + 8*hi + r, uniform idx per half-wave
            const size_t row = waveRow + t * 16 + hi * 8 + r;
            if (mr == 0) idx_out[row] = (float)i;
            const float4* es = (const float4*)(emb + (size_t)i * VQ_D + mr * 16);
            const float4* zs = (const float4*)(z + row * VQ_D + mr * 16);
            float4*       dq = (float4*)(zq + row * VQ_D + mr * 16);
            #pragma unroll
            for (int j = 0; j < 4; ++j) {
                float4 e4 = es[j], z4 = zs[j];
                dq[j] = e4;
                float dx = e4.x - z4.x, dy = e4.y - z4.y;
                float dz = e4.z - z4.z, dw = e4.w - z4.w;
                lsum += dx*dx + dy*dy + dz*dz + dw*dw;
            }
        }
    }
    #pragma unroll
    for (int off = 16; off >= 1; off >>= 1) lsum += __shfl_xor(lsum, off, 32);
    if (lane == 0) partials[blockIdx.x * 8 + wave] = lsum;
}

// ---------- kernel 3: deterministic loss finalize ----------
__global__ __launch_bounds__(32) void vq_loss_finalize(const float* __restrict__ partials,
                                                       float* __restrict__ out_loss) {
    int lane = threadIdx.x;
    float s = 0.0f;
    for (int k = lane; k < N_PARTIALS; k += 32) s += partials[k];
    #pragma unroll
    for (int off = 16; off >= 1; off >>= 1) s += __shfl_xor(s, off, 32);
    if (lane == 0)
        *out_loss = s * (1.0f + VQ_BETA) / (float)((size_t)VQ_N * VQ_D);
}

extern "C" void kernel_launch(void* const* d_in, const int* in_sizes, int n_in,
                              void* d_out, int out_size, void* d_ws, size_t ws_size,
                              hipStream_t stream) {
    const float* z   = (const float*)d_in[0];   // [32,1024,256] f32
    const float* emb = (const float*)d_in[1];   // [8192,256]    f32

    float* out     = (float*)d_out;
    float* zq      = out;                                   // N*D floats
    float* idx_out = out + (size_t)VQ_N * VQ_D;             // N floats (int values)
    float* loss    = out + (size_t)VQ_N * VQ_D + VQ_N;      // 1 float

    float* e2s      = (float*)d_ws;                         // V floats
    float* partials = e2s + VQ_V;                           // N_PARTIALS floats

    vq_e2_kernel<<<VQ_V / 8, 256, 0, stream>>>(emb, e2s);
    vq_main_kernel<<<MAIN_GRID, WG_THREADS, 0, stream>>>(z, emb, e2s, zq, idx_out, partials);
    vq_loss_finalize<<<1, 32, 0, stream>>>(partials, loss);
}
